// OneShotRNN_7662221656299
// MI455X (gfx1250) — matrix operand order repaired and checked by reading, not compiled, not run on hardware
//
#include <hip/hip_runtime.h>
#include <hip/hip_bf16.h>

// Problem constants (fixed by the reference)
#define B_SZ   1024
#define T_SZ   256
#define D_SZ   64
#define U_SZ   512
#define G_SZ   2048      // 4*U
#define K1     576       // U + D   (stacked rows: [Uh1 ; W1])
#define K2     1024      // U + U   (stacked rows: [W2 ; Uh2])
#define KT1    18        // K1/32
#define KT2    32        // K2/32
#define NT     128       // G/16 gate-column tiles
#define STR1   584       // LDS stride (halves); 584 % 128 == 72 -> conflict-free
#define STR2   1096      // 1096 % 128 == 72
#define THREADS 512
#define WAVES   16
#define BTILE   16

typedef _Float16 v16h __attribute__((ext_vector_type(16)));
typedef _Float16 v8h  __attribute__((ext_vector_type(8)));
typedef float    v8f  __attribute__((ext_vector_type(8)));

union V16 { v16h v; v8h h[2]; };

__device__ __forceinline__ float fast_tanh(float x) {
  float y;
  asm("v_tanh_f32 %0, %1" : "=v"(y) : "v"(x));
  return y;
}
__device__ __forceinline__ float sigm(float x) {
  return __builtin_fmaf(0.5f, fast_tanh(0.5f * x), 0.5f);
}
__device__ __forceinline__ v8f zero8() {
  v8f z;
#pragma unroll
  for (int j = 0; j < 8; ++j) z[j] = 0.0f;
  return z;
}

// ---------------------------------------------------------------------------
// Weight prep: convert fp32 weights to f16 and swizzle into this kernel's contiguous-k WMMA
// B-fragment layout:  element index e = ((kt*NT + nt)*32 + lane)*16 + i
//   N = nt*16 + (lane&15);   K = kt*32 + (lane>>4)*16 + i
// Layer 1 stack: rows [0,512) = Uh1 (pairs with h1), rows [512,576) = W1 (x).
// Layer 2 stack: rows [0,512) = W2  (pairs with h1), rows [512,1024) = Uh2 (h2).
// ---------------------------------------------------------------------------
__global__ __launch_bounds__(256) void prep_w1(const float* __restrict__ W1,
                                               const float* __restrict__ Uh1,
                                               _Float16* __restrict__ w1s) {
  int e = blockIdx.x * 256 + threadIdx.x;
  if (e >= KT1 * NT * 512) return;
  int i    = e & 15;
  int lane = (e >> 4) & 31;
  int nt   = (e >> 9) & (NT - 1);
  int kt   = e >> 16;
  int n = nt * 16 + (lane & 15);
  int r = kt * 32 + ((lane >> 4) << 4) + i;
  float v = (r < U_SZ) ? Uh1[(size_t)r * G_SZ + n]
                       : W1[(size_t)(r - U_SZ) * G_SZ + n];
  w1s[e] = (_Float16)v;
}

__global__ __launch_bounds__(256) void prep_w2(const float* __restrict__ W2,
                                               const float* __restrict__ Uh2,
                                               _Float16* __restrict__ w2s) {
  int e = blockIdx.x * 256 + threadIdx.x;
  if (e >= KT2 * NT * 512) return;
  int i    = e & 15;
  int lane = (e >> 4) & 31;
  int nt   = (e >> 9) & (NT - 1);
  int kt   = e >> 16;
  int n = nt * 16 + (lane & 15);
  int r = kt * 32 + ((lane >> 4) << 4) + i;
  float v = (r < U_SZ) ? W2[(size_t)r * G_SZ + n]
                       : Uh2[(size_t)(r - U_SZ) * G_SZ + n];
  w2s[e] = (_Float16)v;
}

// ---------------------------------------------------------------------------
// Persistent LSTM: one block per 16-row batch tile, loops all T steps.
// 16 waves; wave w owns gate-unit tiles u = {w, w+16} for BOTH layers, so the
// cell state c1/c2 lives in v8f registers for the whole time loop.
// ---------------------------------------------------------------------------
__global__ __launch_bounds__(THREADS) void lstm_persist(
    const float* __restrict__ x,
    const float* __restrict__ b1, const float* __restrict__ b2,
    const float* __restrict__ Wd, const float* __restrict__ bd,
    const _Float16* __restrict__ w1s, const _Float16* __restrict__ w2s,
    float* __restrict__ out) {
  __shared__ _Float16 act1[BTILE * STR1];  // cols [0,512)=h1, [512,576)=x_t
  __shared__ _Float16 act2[BTILE * STR2];  // cols [0,512)=h1, [512,1024)=h2

  const int tid  = threadIdx.x;
  const int wid  = tid >> 5;
  const int lane = tid & 31;
  const int ln   = lane & 15;   // M for A-frag / N for B-frag & C tile
  const int g    = lane >> 4;   // lane group
  const int b0   = blockIdx.x * BTILE;
  const float* xblk = x + (size_t)b0 * T_SZ * D_SZ;

  // zero-init LDS h state
  for (int idx = tid; idx < BTILE * STR1; idx += THREADS) act1[idx] = (_Float16)0.0f;
  for (int idx = tid; idx < BTILE * STR2; idx += THREADS) act2[idx] = (_Float16)0.0f;

  // per-wave biases for the two owned u-tiles (gate order i,f,g,o)
  float bb1[2][4], bb2[2][4];
#pragma unroll
  for (int s = 0; s < 2; ++s) {
    int u = wid + s * WAVES;
#pragma unroll
    for (int gate = 0; gate < 4; ++gate) {
      bb1[s][gate] = b1[gate * U_SZ + u * 16 + ln];
      bb2[s][gate] = b2[gate * U_SZ + u * 16 + ln];
    }
  }
  v8f c1v[2], c2v[2];
  c1v[0] = zero8(); c1v[1] = zero8();
  c2v[0] = zero8(); c2v[1] = zero8();

  __syncthreads();

  const v8h* a1row = (const v8h*)(act1 + ln * STR1);
  const v8h* a2row = (const v8h*)(act2 + ln * STR2);
  const v8h* w1v   = (const v8h*)w1s;
  const v8h* w2v   = (const v8h*)w2s;

  for (int t = 0; t < T_SZ; ++t) {
    // (a) stage x_t tile (f32 -> f16) into act1 cols [512,576)
    for (int e = tid; e < BTILE * D_SZ; e += THREADS) {
      int m = e >> 6, d = e & 63;
      act1[m * STR1 + U_SZ + d] = (_Float16)xblk[((size_t)m * T_SZ + t) * D_SZ + d];
    }
    __syncthreads();  // (b)

    // ---- layer 1: z = [h1|x] @ [Uh1;W1] ----
    float hn1[2][8];
#pragma unroll
    for (int s = 0; s < 2; ++s) {
      const int u = wid + s * WAVES;
      v8f acc[4];
      acc[0] = zero8(); acc[1] = zero8(); acc[2] = zero8(); acc[3] = zero8();
      for (int kt = 0; kt < KT1; ++kt) {
        V16 a;
        a.h[0] = a1row[kt * 4 + 2 * g];
        a.h[1] = a1row[kt * 4 + 2 * g + 1];
#pragma unroll
        for (int gate = 0; gate < 4; ++gate) {
          const v8h* bp = w1v + ((size_t)(kt * NT + gate * 32 + u) * 32 + lane) * 2;
          V16 bf; bf.h[0] = bp[0]; bf.h[1] = bp[1];
          acc[gate] = __builtin_amdgcn_wmma_f32_16x16x32_f16(
              false, a.v, false, bf.v, (short)0, acc[gate], false, false);
        }
      }
#pragma unroll
      for (int j = 0; j < 8; ++j) {
        float zi = acc[0][j] + bb1[s][0];
        float zf = acc[1][j] + bb1[s][1];
        float zg = acc[2][j] + bb1[s][2];
        float zo = acc[3][j] + bb1[s][3];
        float cn = sigm(zf) * c1v[s][j] + sigm(zi) * fast_tanh(zg);
        c1v[s][j] = cn;
        hn1[s][j] = sigm(zo) * fast_tanh(cn);
      }
    }
    __syncthreads();  // (e) all layer-1 reads of act1 complete
#pragma unroll
    for (int s = 0; s < 2; ++s) {
      int col = (wid + s * WAVES) * 16 + ln;
#pragma unroll
      for (int j = 0; j < 8; ++j) {
        _Float16 hv = (_Float16)hn1[s][j];
        int M = g * 8 + j;
        act1[M * STR1 + col] = hv;   // h1 for next step's layer 1
        act2[M * STR2 + col] = hv;   // h1 input for layer 2 now
      }
    }
    __syncthreads();  // (g) h1 visible

    // ---- layer 2: z = [h1|h2] @ [W2;Uh2] ----
    float hn2[2][8];
#pragma unroll
    for (int s = 0; s < 2; ++s) {
      const int u = wid + s * WAVES;
      v8f acc[4];
      acc[0] = zero8(); acc[1] = zero8(); acc[2] = zero8(); acc[3] = zero8();
      for (int kt = 0; kt < KT2; ++kt) {
        V16 a;
        a.h[0] = a2row[kt * 4 + 2 * g];
        a.h[1] = a2row[kt * 4 + 2 * g + 1];
#pragma unroll
        for (int gate = 0; gate < 4; ++gate) {
          const v8h* bp = w2v + ((size_t)(kt * NT + gate * 32 + u) * 32 + lane) * 2;
          V16 bf; bf.h[0] = bp[0]; bf.h[1] = bp[1];
          acc[gate] = __builtin_amdgcn_wmma_f32_16x16x32_f16(
              false, a.v, false, bf.v, (short)0, acc[gate], false, false);
        }
      }
#pragma unroll
      for (int j = 0; j < 8; ++j) {
        float zi = acc[0][j] + bb2[s][0];
        float zf = acc[1][j] + bb2[s][1];
        float zg = acc[2][j] + bb2[s][2];
        float zo = acc[3][j] + bb2[s][3];
        float cn = sigm(zf) * c2v[s][j] + sigm(zi) * fast_tanh(zg);
        c2v[s][j] = cn;
        hn2[s][j] = sigm(zo) * fast_tanh(cn);
      }
    }
    __syncthreads();  // (j) all layer-2 reads of act2 complete
#pragma unroll
    for (int s = 0; s < 2; ++s) {
      int col = (wid + s * WAVES) * 16 + ln;
#pragma unroll
      for (int j = 0; j < 8; ++j) {
        act2[(g * 8 + j) * STR2 + U_SZ + col] = (_Float16)hn2[s][j];
      }
    }
    // next iteration's barriers (b)/(e)/(g) order these writes vs. readers
  }

  __syncthreads();
  // Dense head: out[b] = h2_final[b,:] @ Wd + bd   (deterministic serial dot)
  if (tid < BTILE) {
    float sacc = bd[0];
    for (int uu = 0; uu < U_SZ; ++uu)
      sacc += (float)act2[tid * STR2 + U_SZ + uu] * Wd[uu];
    out[b0 + tid] = sacc;
  }
}

extern "C" void kernel_launch(void* const* d_in, const int* in_sizes, int n_in,
                              void* d_out, int out_size, void* d_ws, size_t ws_size,
                              hipStream_t stream) {
  (void)in_sizes; (void)n_in; (void)out_size; (void)ws_size;
  const float* x   = (const float*)d_in[0];
  const float* W1  = (const float*)d_in[1];
  const float* Uh1 = (const float*)d_in[2];
  const float* b1  = (const float*)d_in[3];
  const float* W2  = (const float*)d_in[4];
  const float* Uh2 = (const float*)d_in[5];
  const float* b2  = (const float*)d_in[6];
  const float* Wd  = (const float*)d_in[7];
  const float* bd  = (const float*)d_in[8];

  _Float16* w1s = (_Float16*)d_ws;                       // 1,179,648 halves
  _Float16* w2s = w1s + (size_t)KT1 * NT * 512;          // 2,097,152 halves

  const int N1 = KT1 * NT * 512;
  const int N2 = KT2 * NT * 512;
  prep_w1<<<(N1 + 255) / 256, 256, 0, stream>>>(W1, Uh1, w1s);
  prep_w2<<<(N2 + 255) / 256, 256, 0, stream>>>(W2, Uh2, w2s);
  lstm_persist<<<B_SZ / BTILE, THREADS, 0, stream>>>(x, b1, b2, Wd, bd,
                                                     w1s, w2s, (float*)d_out);
}
